// L2Loss_with_penality_16587163698096
// MI455X (gfx1250) — compile-verified, matrix operand order checked by reading, not run on hardware
//
#include <hip/hip_runtime.h>

#define EPSV 1e-6f
#define NBINS 65536
#define TILE 1024       // 256 threads * 4 floats per staged tile
#define HOTLO 0x3B80u   // coarse bin of p = 2^-8; bins [HOTLO, 0x3F80) privatized
#define HOTN 1024       // 0x3F7F - 0x3B80 + 1 (max coarse bin for p < 1.0)

typedef __attribute__((ext_vector_type(16))) _Float16 v16h;
typedef __attribute__((ext_vector_type(8)))  float    v8f;

#if __has_builtin(__builtin_amdgcn_global_load_async_to_lds_b128) && \
    __has_builtin(__builtin_amdgcn_s_wait_asynccnt)
#define USE_ASYNC_LDS 1
// Parameter types per hipcc's diagnostic:
//   '__attribute__((__vector_size__(4 * sizeof(int)))) int __device__ *'
typedef int v4i_vs __attribute__((vector_size(4 * sizeof(int))));
typedef __attribute__((address_space(1))) v4i_vs* gptr_b128;  // global (AS1)
typedef __attribute__((address_space(3))) v4i_vs* lptr_b128;  // LDS (AS3)
#else
#define USE_ASYNC_LDS 0
#endif

__device__ __forceinline__ float clampp(float x) {
  return fminf(fmaxf(x, EPSV), 1.0f - EPSV);
}
__device__ __forceinline__ float wgt(float a) { return (a < 1.0f) ? 20.0f : a; }

// -------------------- zero scratch --------------------
__global__ void k_zero(float* p, int n) {
  int i = blockIdx.x * blockDim.x + threadIdx.x;
  int s = gridDim.x * blockDim.x;
  for (; i < n; i += s) p[i] = 0.0f;
}

// ------ pass 1: coarse histogram over (bits>>16) + mse partials ------
// Double-buffered async global->LDS staging (ASYNCcnt) + LDS-privatized
// hot-range histogram (ds_add_f32) to kill L2 same-address atomic serialization.
__global__ void k_hist_coarse(const float* __restrict__ pred,
                              const float* __restrict__ actual,
                              float* __restrict__ hist,
                              float* __restrict__ scal, int n) {
  __shared__ float sp[2][TILE];
  __shared__ float sa[2][TILE];
  __shared__ float lh[HOTN];
  int t = threadIdx.x;
  for (int j = t; j < HOTN; j += 256) lh[j] = 0.0f;
  __syncthreads();

  float msum = 0.0f;
  long long nfull = (long long)(n / TILE) * TILE;
  long long tstep = (long long)gridDim.x * TILE;
  long long base0 = (long long)blockIdx.x * TILE;

  // stage first tile
  if (base0 < nfull) {
    int g = (int)base0 + t * 4;
#if USE_ASYNC_LDS
    __builtin_amdgcn_global_load_async_to_lds_b128(
        (gptr_b128)(void*)(pred + g), (lptr_b128)(void*)(&sp[0][t * 4]), 0, 0);
    __builtin_amdgcn_global_load_async_to_lds_b128(
        (gptr_b128)(void*)(actual + g), (lptr_b128)(void*)(&sa[0][t * 4]), 0, 0);
#else
    *(float4*)(&sp[0][t * 4]) = *(const float4*)(pred + g);
    *(float4*)(&sa[0][t * 4]) = *(const float4*)(actual + g);
#endif
  }

  int buf = 0;
  for (long long base = base0; base < nfull; base += tstep) {
    long long next = base + tstep;
    bool more = next < nfull;  // block-uniform
    if (more) {
      int g = (int)next + t * 4;
#if USE_ASYNC_LDS
      __builtin_amdgcn_global_load_async_to_lds_b128(
          (gptr_b128)(void*)(pred + g),
          (lptr_b128)(void*)(&sp[buf ^ 1][t * 4]), 0, 0);
      __builtin_amdgcn_global_load_async_to_lds_b128(
          (gptr_b128)(void*)(actual + g),
          (lptr_b128)(void*)(&sa[buf ^ 1][t * 4]), 0, 0);
#else
      *(float4*)(&sp[buf ^ 1][t * 4]) = *(const float4*)(pred + g);
      *(float4*)(&sa[buf ^ 1][t * 4]) = *(const float4*)(actual + g);
#endif
    }
#if USE_ASYNC_LDS
    // async loads complete in order: keep the 2 just-issued ops in flight,
    // wait only for the current buffer's 2 ops
    if (more) {
      __builtin_amdgcn_s_wait_asynccnt(2);
    } else {
      __builtin_amdgcn_s_wait_asynccnt(0);
    }
#endif
    // each thread consumes only the 16B it staged itself -> no barrier needed
#pragma unroll
    for (int k = 0; k < 4; ++k) {
      float p = clampp(sp[buf][t * 4 + k]);
      float a = sa[buf][t * 4 + k];
      unsigned bin = __float_as_uint(p) >> 16;
      float w = wgt(a);
      if (bin >= HOTLO) {
        __hip_atomic_fetch_add(&lh[bin - HOTLO], w, __ATOMIC_RELAXED,
                               __HIP_MEMORY_SCOPE_WORKGROUP);
      } else {
        unsafeAtomicAdd(&hist[bin], w);
      }
      float d = p - a;
      msum += d * d;
    }
    buf ^= 1;
  }

  // tail elements [nfull, n): block 0 handles directly from global
  if (blockIdx.x == 0) {
    for (long long i = nfull + t; i < n; i += 256) {
      float p = clampp(pred[i]);
      float a = actual[i];
      unsafeAtomicAdd(&hist[__float_as_uint(p) >> 16], wgt(a));
      float d = p - a;
      msum += d * d;
    }
  }

  // flush privatized hot bins
  __syncthreads();
  for (int j = t; j < HOTN; j += 256) {
    float v = lh[j];
    if (v != 0.0f) unsafeAtomicAdd(&hist[HOTLO + j], v);
  }

  for (int off = 16; off > 0; off >>= 1) msum += __shfl_xor(msum, off, 32);
  if ((t & 31) == 0) unsafeAtomicAdd(&scal[0], msum);
}

// ------ single-block descending scan to locate crossing bin ------
// mode 0: coarse (computes W, T; stores T->scal[1], A1->scal[2], b1->scal[3])
// mode 1: fine   (uses T, A1, b1; stores thr->scal[4])
__global__ void k_scan(const float* __restrict__ hist, float* scal, int mode) {
  __shared__ float csum[256];
  int t = threadIdx.x;
  float s = 0.0f;
  for (int j = 0; j < 256; ++j) s += hist[t * 256 + j];
  csum[t] = s;
  __syncthreads();
  if (t == 0) {
    float T, run;
    if (mode == 0) {
      float W = 0.0f;
      for (int c = 0; c < 256; ++c) W += csum[c];
      T = 0.04f * W;
      run = 0.0f;
      scal[1] = T;
    } else {
      T = scal[1];
      run = scal[2];
    }
    int fb = -1;
    for (int c = 255; c >= 0 && fb < 0; --c) {
      if (run + csum[c] > T) {
        int lo = c * 256;
        for (int b = lo + 255; b >= lo; --b) {
          float h = hist[b];
          if (run + h > T) { fb = b; break; }
          run += h;
        }
        // rounding miss inside chunk: continue with next chunk
      } else {
        run += csum[c];
      }
    }
    if (fb < 0) fb = 0;
    if (mode == 0) {
      scal[2] = run;                         // weight strictly above bin b1
      ((unsigned*)scal)[3] = (unsigned)fb;   // coarse bin index
    } else {
      unsigned b1 = ((const unsigned*)scal)[3];
      scal[4] = __uint_as_float((b1 << 16) | (unsigned)fb);  // exact thr
    }
  }
}

// ------ pass 2: fine histogram over (bits & 0xFFFF) inside coarse bin b1 ------
__global__ void k_hist_fine(const float* __restrict__ pred,
                            const float* __restrict__ actual,
                            float* __restrict__ hist,
                            const float* __restrict__ scal, int n) {
  unsigned b1 = ((const unsigned*)scal)[3];
  int i = blockIdx.x * blockDim.x + threadIdx.x;
  int stride = gridDim.x * blockDim.x;
  int i4 = i * 4, s4 = stride * 4;
  for (; i4 + 3 < n; i4 += s4) {
    __builtin_prefetch(pred + i4 + s4, 0, 0);    // global_prefetch_b8
    __builtin_prefetch(actual + i4 + s4, 0, 0);
    float4 p4 = *(const float4*)(pred + i4);
    float4 a4 = *(const float4*)(actual + i4);
    float pk[4] = {p4.x, p4.y, p4.z, p4.w};
    float ak[4] = {a4.x, a4.y, a4.z, a4.w};
#pragma unroll
    for (int k = 0; k < 4; ++k) {
      unsigned bits = __float_as_uint(clampp(pk[k]));
      if ((bits >> 16) == b1)
        unsafeAtomicAdd(&hist[bits & 0xFFFFu], wgt(ak[k]));
    }
  }
  int r = (n & ~3) + i;  // scalar tail
  if (r < n) {
    unsigned bits = __float_as_uint(clampp(pred[r]));
    if ((bits >> 16) == b1)
      unsafeAtomicAdd(&hist[bits & 0xFFFFu], wgt(actual[r]));
  }
}

// ------ pass 3: masked -log sum + count; count reduced exactly via WMMA ------
__global__ void k_final(const float* __restrict__ pred,
                        const float* __restrict__ actual,
                        float* __restrict__ scal, int n) {
  float thr = scal[4];
  float ex = 0.0f;
  int c = 0;
  int i = blockIdx.x * blockDim.x + threadIdx.x;
  int stride = gridDim.x * blockDim.x;
  int i4 = i * 4, s4 = stride * 4;
  for (; i4 + 3 < n; i4 += s4) {
    __builtin_prefetch(pred + i4 + s4, 0, 0);
    __builtin_prefetch(actual + i4 + s4, 0, 0);
    float4 p4 = *(const float4*)(pred + i4);
    float4 a4 = *(const float4*)(actual + i4);
    float pk[4] = {p4.x, p4.y, p4.z, p4.w};
    float ak[4] = {a4.x, a4.y, a4.z, a4.w};
#pragma unroll
    for (int k = 0; k < 4; ++k) {
      float p = clampp(pk[k]);
      if (ak[k] == 0.0f && p > thr) {
        ex += -logf(1.0f - p + thr);
        ++c;
      }
    }
  }
  int r = (n & ~3) + i;
  if (r < n) {
    float p = clampp(pred[r]);
    if (actual[r] == 0.0f && p > thr) {
      ex += -logf(1.0f - p + thr);
      ++c;
    }
  }
  // extra_loss: float shuffle reduction
  for (int off = 16; off > 0; off >>= 1) ex += __shfl_xor(ex, off, 32);
  // count: exact wave reduction via v_wmma_f32_16x16x32_f16.
  // Low 10 bits per lane (<=1023, exact in f16) go into A; B = ones; f32 acc.
  // Sum over all D rows == sum of every A entry == sum of lane counts.
  int clo = c & 1023;
  int chi = c >> 10;  // almost always 0; reduced by shuffle for robustness
  v16h A = {};
  v16h B;
#pragma unroll
  for (int k = 0; k < 16; ++k) B[k] = (_Float16)1.0f;
  A[0] = (_Float16)(float)clo;
  v8f C = {};
  v8f D = __builtin_amdgcn_wmma_f32_16x16x32_f16(
      /*neg_a=*/false, A, /*neg_b=*/false, B,
      /*c_mod=*/(short)0, C, /*reuse_a=*/false, /*reuse_b=*/false);
  float cw = D[0] + D[1] + D[2] + D[3] + D[4] + D[5] + D[6] + D[7];
  cw += __shfl_xor(cw, 16, 32);  // rows 0-7 half + rows 8-15 half
  float fhi = (float)chi;
  for (int off = 16; off > 0; off >>= 1) fhi += __shfl_xor(fhi, off, 32);
  float cnt = cw + 1024.0f * fhi;
  if ((threadIdx.x & 31) == 0) {
    unsafeAtomicAdd(&scal[5], ex);
    unsafeAtomicAdd(&scal[6], cnt);
  }
}

// -------------------- compose scalar output --------------------
__global__ void k_finalize(const float* __restrict__ scal,
                           float* __restrict__ out, int n) {
  out[0] = scal[0] / (float)n + scal[5] / scal[6];
}

extern "C" void kernel_launch(void* const* d_in, const int* in_sizes, int n_in,
                              void* d_out, int out_size, void* d_ws, size_t ws_size,
                              hipStream_t stream) {
  const float* pred = (const float*)d_in[0];
  const float* actual = (const float*)d_in[1];
  int n = in_sizes[0];
  float* hist = (float*)d_ws;       // 65536 f32, reused coarse -> fine
  float* scal = hist + NBINS;       // 16 f32 scalars
  float* out = (float*)d_out;

  int tiles = (n + TILE - 1) / TILE;
  int cblocks = tiles < 512 ? (tiles > 0 ? tiles : 1) : 512;

  k_zero<<<256, 256, 0, stream>>>(hist, NBINS + 16);
  k_hist_coarse<<<cblocks, 256, 0, stream>>>(pred, actual, hist, scal, n);
  k_scan<<<1, 256, 0, stream>>>(hist, scal, 0);
  k_zero<<<256, 256, 0, stream>>>(hist, NBINS);
  k_hist_fine<<<2048, 256, 0, stream>>>(pred, actual, hist, scal, n);
  k_scan<<<1, 256, 0, stream>>>(hist, scal, 1);
  k_final<<<2048, 256, 0, stream>>>(pred, actual, scal, n);
  k_finalize<<<1, 1, 0, stream>>>(scal, out, n);
}
